// SpeechT5CrossAttention_884763263359
// MI455X (gfx1250) — matrix operand order repaired and checked by reading, not compiled, not run on hardware
//
#include <hip/hip_runtime.h>

typedef __attribute__((ext_vector_type(16))) __bf16 v16bf;
typedef __attribute__((ext_vector_type(8)))  float  v8f;

#define BB 4
#define TT 1024
#define SS 1024
#define EE 768
#define HH 12
#define DD 64
#define QSCALE 0.125f  // 64^-0.5

static __device__ __forceinline__ v8f wmma_bf16(v16bf a, v16bf b, v8f c) {
  // D = A(16x32 bf16) x B(32x16 bf16) + C(16x16 f32)
  return __builtin_amdgcn_wmma_f32_16x16x32_bf16(false, a, false, b, (short)0, c,
                                                 false, false);
}

// ---------------------------------------------------------------------------
// fp32 -> bf16 weight conversion
// ---------------------------------------------------------------------------
__global__ void cvt_f32_bf16(const float* __restrict__ src,
                             __bf16* __restrict__ dst, int n) {
  int i = blockIdx.x * blockDim.x + threadIdx.x;
  int stride = gridDim.x * blockDim.x;
  for (; i < n; i += stride) dst[i] = (__bf16)src[i];
}

// ---------------------------------------------------------------------------
// QKV projection: Y = X @ W^T + b  (one 16x16 output tile per wave)
// mode 0: Q  -> [B,H,T,D] bf16, scaled by QSCALE
// mode 1: K  -> [B,H,S,D] bf16
// mode 2: V  -> [B,H,D,S] bf16 (transposed for P@V B-fragments)
// ---------------------------------------------------------------------------
__global__ __launch_bounds__(256)
void proj_kernel(const float* __restrict__ X, const __bf16* __restrict__ Wbf,
                 const float* __restrict__ bias, __bf16* __restrict__ Y, int mode) {
  const int wave = (blockIdx.x * blockDim.x + threadIdx.x) >> 5;
  const int lane = threadIdx.x & 31;
  const int row0 = (wave / (EE / 16)) * 16;
  const int col0 = (wave % (EE / 16)) * 16;
  const int m    = lane & 15;   // A row / B col / C col
  const int half = lane >> 4;

  v8f c = {};
  const float*  xr = X   + (size_t)(row0 + m) * EE;
  const __bf16* wr = Wbf + (size_t)(col0 + m) * EE + half * 16;
  for (int k0 = 0; k0 < EE; k0 += 32) {
    v16bf a, b;
    const float* xa = xr + k0 + half * 16;
#pragma unroll
    for (int i = 0; i < 8; ++i) {
      a[i]     = (__bf16)xa[i];
      a[8 + i] = (__bf16)xa[8 + i];
    }
    b = *(const v16bf*)(wr + k0);
    c = wmma_bf16(a, b, c);
  }

  const float bias_v = bias[col0 + m];
  const int   col    = col0 + m;
  const int   h      = col >> 6;   // /64
  const int   d      = col & 63;
#pragma unroll
  for (int r = 0; r < 8; ++r) {
    const int rr  = row0 + half * 8 + r;
    const int b_i = rr >> 10;      // /1024
    const int t   = rr & 1023;
    float y = c[r] + bias_v;
    if (mode == 0) y *= QSCALE;
    if (mode == 2)
      Y[(((size_t)b_i * HH + h) * DD + d) * SS + t] = (__bf16)y;
    else
      Y[(((size_t)b_i * HH + h) * TT + t) * DD + d] = (__bf16)y;
  }
}

// ---------------------------------------------------------------------------
// Flash attention: one wave per (t-tile of 16, b, h). Wave order is
// t-tile-major so concurrent workgroups reuse the same position_bias rows
// out of L2 (pb is the dominant 256MB tensor).
// ---------------------------------------------------------------------------
__global__ __launch_bounds__(256)
void attn_kernel(const __bf16* __restrict__ qb, const __bf16* __restrict__ kb,
                 const __bf16* __restrict__ vtb, const float* __restrict__ pb,
                 __bf16* __restrict__ attn) {
  __shared__ float  qtile[8][16][DD];   // 32 KB: fp32 q rows for pb dot
  __shared__ __bf16 plds[8][16 * 32];   //  8 KB: probs re-layout buffer

  const int w      = threadIdx.x >> 5;
  const int lane   = threadIdx.x & 31;
  const int wg     = blockIdx.x * 8 + w;
  const int t_tile = wg / (BB * HH);
  const int bh     = wg % (BB * HH);
  const int bi     = bh / HH, h = bh % HH;
  const int t0     = t_tile * 16;
  const int m      = lane & 15;
  const int half   = lane >> 4;

  const __bf16* qrow_base = qb  + ((size_t)bi * HH + h) * TT * DD;
  const __bf16* krow_base = kb  + ((size_t)bi * HH + h) * SS * DD;
  const __bf16* vt_base   = vtb + ((size_t)bi * HH + h) * DD * SS;

  // stage the (scaled) fp32 q tile in LDS for the position-bias contraction
  for (int idx = lane; idx < 16 * DD; idx += 32) {
    const int rr = idx >> 6, dd = idx & 63;
    qtile[w][rr][dd] = (float)qrow_base[(size_t)(t0 + rr) * DD + dd];
  }

  // Q A-fragments for d=0..31 and d=32..63
  v16bf aq0, aq1;
  {
    const __bf16* qr = qrow_base + (size_t)(t0 + m) * DD;
#pragma unroll
    for (int i = 0; i < 8; ++i) {
      aq0[i]     = qr[half * 16 + i];
      aq0[8 + i] = qr[half * 16 + 8 + i];
      aq1[i]     = qr[32 + half * 16 + i];
      aq1[8 + i] = qr[32 + half * 16 + 8 + i];
    }
  }

  float mrun[8], lrun[8];
  v8f o0 = {}, o1 = {}, o2 = {}, o3 = {};
#pragma unroll
  for (int r = 0; r < 8; ++r) { mrun[r] = -1e30f; lrun[r] = 0.f; }

  __syncthreads();  // qtile ready

  for (int s0 = 0; s0 < SS; s0 += 32) {
    // ---- scores = Q K^T for two 16-wide column tiles -----------------
    v8f c0 = {}, c1 = {};
    {
      const __bf16* k0p = krow_base + (size_t)(s0 + m) * DD + half * 16;
      const __bf16* k1p = k0p + 16 * DD;
      v16bf b00 = *(const v16bf*)(k0p);
      v16bf b01 = *(const v16bf*)(k0p + 32);
      v16bf b10 = *(const v16bf*)(k1p);
      v16bf b11 = *(const v16bf*)(k1p + 32);
      c0 = wmma_bf16(aq0, b00, c0);
      c0 = wmma_bf16(aq1, b01, c0);
      c1 = wmma_bf16(aq0, b10, c1);
      c1 = wmma_bf16(aq1, b11, c1);
    }

    // ---- scores += einsum(q[t,:], pb[t,s,:])  (VALU, fp32, bw-bound) --
#pragma unroll
    for (int r = 0; r < 8; ++r) {
      const int mrow = half * 8 + r;
      const int tg   = t0 + mrow;
      const float4* p0 = (const float4*)(pb + ((size_t)tg * SS + (s0 + m)) * DD);
      const float4* p1 = p0 + (16 * DD) / 4;
      const float4* qv = (const float4*)(&qtile[w][mrow][0]);
      if (s0 + 32 < SS)
        __builtin_prefetch(pb + ((size_t)tg * SS + (s0 + 32 + m)) * DD, 0, 1);
      float a0 = 0.f, a1 = 0.f;
#pragma unroll
      for (int d4 = 0; d4 < DD / 4; ++d4) {
        const float4 q4 = qv[d4];
        const float4 x0 = p0[d4];
        const float4 x1 = p1[d4];
        a0 += q4.x * x0.x + q4.y * x0.y + q4.z * x0.z + q4.w * x0.w;
        a1 += q4.x * x1.x + q4.y * x1.y + q4.z * x1.z + q4.w * x1.w;
      }
      c0[r] += a0;
      c1[r] += a1;
    }

    // ---- online softmax over this 32-column block --------------------
#pragma unroll
    for (int r = 0; r < 8; ++r) {
      const float v0 = c0[r], v1 = c1[r];
      float bm = fmaxf(v0, v1);
#pragma unroll
      for (int off = 1; off < 16; off <<= 1)
        bm = fmaxf(bm, __shfl_xor(bm, off, 16));
      const float nm = fmaxf(mrun[r], bm);
      const float sc = __expf(mrun[r] - nm);
      const float p0 = __expf(v0 - nm);
      const float p1 = __expf(v1 - nm);
      float ps = p0 + p1;
#pragma unroll
      for (int off = 1; off < 16; off <<= 1)
        ps += __shfl_xor(ps, off, 16);
      lrun[r] = lrun[r] * sc + ps;
      mrun[r] = nm;
      o0[r] *= sc; o1[r] *= sc; o2[r] *= sc; o3[r] *= sc;
      const int mrow = half * 8 + r;
      plds[w][mrow * 32 + m]      = (__bf16)p0;
      plds[w][mrow * 32 + 16 + m] = (__bf16)p1;
    }
    __syncthreads();  // probs visible (uniform trip count across all waves)

    // ---- P @ V  (re-layout probs C->A via LDS) -----------------------
    v16bf ap;
#pragma unroll
    for (int i = 0; i < 8; ++i) {
      ap[i]     = plds[w][m * 32 + half * 16 + i];
      ap[8 + i] = plds[w][m * 32 + half * 16 + 8 + i];
    }
    const __bf16* v0p = vt_base + (size_t)m * SS + s0 + half * 16;
    v16bf bv0 = *(const v16bf*)(v0p);
    v16bf bv1 = *(const v16bf*)(v0p + 16 * SS);
    v16bf bv2 = *(const v16bf*)(v0p + 32 * SS);
    v16bf bv3 = *(const v16bf*)(v0p + 48 * SS);
    o0 = wmma_bf16(ap, bv0, o0);
    o1 = wmma_bf16(ap, bv1, o1);
    o2 = wmma_bf16(ap, bv2, o2);
    o3 = wmma_bf16(ap, bv3, o3);
  }

  // ---- normalize and store to [B,T,E] bf16 for the output projection ----
#pragma unroll
  for (int r = 0; r < 8; ++r) {
    const int  mrow = half * 8 + r;
    const int  tg   = t0 + mrow;
    const float inv = 1.0f / lrun[r];
    const size_t base = ((size_t)bi * TT + tg) * EE + h * DD;
    attn[base + 0  + m] = (__bf16)(o0[r] * inv);
    attn[base + 16 + m] = (__bf16)(o1[r] * inv);
    attn[base + 32 + m] = (__bf16)(o2[r] * inv);
    attn[base + 48 + m] = (__bf16)(o3[r] * inv);
  }
}

// ---------------------------------------------------------------------------
// Output projection: out[4096,768] = attn_bf16 @ Wo^T + bo (fp32 result)
// ---------------------------------------------------------------------------
__global__ __launch_bounds__(256)
void oproj_kernel(const __bf16* __restrict__ A, const __bf16* __restrict__ Wbf,
                  const float* __restrict__ bias, float* __restrict__ Y) {
  const int wave = (blockIdx.x * blockDim.x + threadIdx.x) >> 5;
  const int lane = threadIdx.x & 31;
  const int row0 = (wave / (EE / 16)) * 16;
  const int col0 = (wave % (EE / 16)) * 16;
  const int m    = lane & 15;
  const int half = lane >> 4;

  v8f c = {};
  const __bf16* ar = A   + (size_t)(row0 + m) * EE;
  const __bf16* wr = Wbf + (size_t)(col0 + m) * EE + half * 16;
  for (int k0 = 0; k0 < EE; k0 += 32) {
    v16bf a, b;
    const __bf16* ax = ar + k0 + half * 16;
#pragma unroll
    for (int i = 0; i < 8; ++i) {
      a[i]     = ax[i];
      a[8 + i] = ax[8 + i];
    }
    b = *(const v16bf*)(wr + k0);
    c = wmma_bf16(a, b, c);
  }
  const float bv = bias[col0 + m];
#pragma unroll
  for (int r = 0; r < 8; ++r) {
    const int rr = row0 + half * 8 + r;
    Y[(size_t)rr * EE + col0 + m] = c[r] + bv;
  }
}

// ---------------------------------------------------------------------------
extern "C" void kernel_launch(void* const* d_in, const int* in_sizes, int n_in,
                              void* d_out, int out_size, void* d_ws, size_t ws_size,
                              hipStream_t stream) {
  const float* hs = (const float*)d_in[0];   // hidden_states  [B,T,E]
  const float* kv = (const float*)d_in[1];   // key_value_states [B,S,E]
  const float* pb = (const float*)d_in[2];   // position_bias [T,S,D]
  const float* Wq = (const float*)d_in[3];
  const float* bq = (const float*)d_in[4];
  const float* Wk = (const float*)d_in[5];
  const float* bk = (const float*)d_in[6];
  const float* Wv = (const float*)d_in[7];
  const float* bv = (const float*)d_in[8];
  const float* Wo = (const float*)d_in[9];
  const float* bo = (const float*)d_in[10];
  float* out = (float*)d_out;

  char* ws = (char*)d_ws;
  const size_t WSZ = (size_t)EE * EE * sizeof(__bf16);            // 1.18 MB
  const size_t HSZ = (size_t)BB * HH * TT * DD * sizeof(__bf16);  // 6 MB
  __bf16* wq_bf = (__bf16*)(ws);
  __bf16* wk_bf = (__bf16*)(ws + WSZ);
  __bf16* wv_bf = (__bf16*)(ws + 2 * WSZ);
  __bf16* wo_bf = (__bf16*)(ws + 3 * WSZ);
  __bf16* q_bf  = (__bf16*)(ws + 4 * WSZ);
  __bf16* k_bf  = (__bf16*)(ws + 4 * WSZ + HSZ);
  __bf16* vt_bf = (__bf16*)(ws + 4 * WSZ + 2 * HSZ);
  __bf16* at_bf = (__bf16*)(ws + 4 * WSZ + 3 * HSZ);

  const int wn = EE * EE;
  cvt_f32_bf16<<<512, 256, 0, stream>>>(Wq, wq_bf, wn);
  cvt_f32_bf16<<<512, 256, 0, stream>>>(Wk, wk_bf, wn);
  cvt_f32_bf16<<<512, 256, 0, stream>>>(Wv, wv_bf, wn);
  cvt_f32_bf16<<<512, 256, 0, stream>>>(Wo, wo_bf, wn);

  // 12288 output tiles each -> 1536 blocks of 8 waves
  proj_kernel<<<1536, 256, 0, stream>>>(hs, wq_bf, bq, q_bf, 0);
  proj_kernel<<<1536, 256, 0, stream>>>(kv, wk_bf, bk, k_bf, 1);
  proj_kernel<<<1536, 256, 0, stream>>>(kv, wv_bf, bv, vt_bf, 2);

  // 64 t-tiles * 48 (b,h) = 3072 waves -> 384 blocks
  attn_kernel<<<384, 256, 0, stream>>>(q_bf, k_bf, vt_bf, pb, at_bf);

  oproj_kernel<<<1536, 256, 0, stream>>>(at_bf, wo_bf, bo, out);
}
